// SlidingWindowAttention_21835613733529
// MI455X (gfx1250) — matrix operand order repaired and checked by reading, not compiled, not run on hardware
//
#include <hip/hip_runtime.h>

typedef __attribute__((ext_vector_type(16))) _Float16 v16h;
typedef __attribute__((ext_vector_type(8)))  _Float16 v8h;
typedef __attribute__((ext_vector_type(8)))  float    v8f;
typedef __attribute__((ext_vector_type(4)))  int      i4;

#define S_LEN   4096
#define DM      1024
#define NH      16
#define DKH     64
#define WINDOW  256

#if defined(__has_builtin)
#if __has_builtin(__builtin_amdgcn_global_load_async_to_lds_b128)
#define ASYNC_LDS 1
#else
#define ASYNC_LDS 0
#endif
#else
#define ASYNC_LDS 0
#endif

__device__ __forceinline__ v8f wmma_f16(v16h a, v16h b, v8f c) {
  // D = A(16x32 f16) * B(32x16 f16) + C(16x16 f32)
  return __builtin_amdgcn_wmma_f32_16x16x32_f16(false, a, false, b, (short)0, c,
                                                false, false);
}

__device__ __forceinline__ v16h cat8(v8h lo, v8h hi) {
  v16h r;
#pragma unroll
  for (int e = 0; e < 8; ++e) { r[e] = lo[e]; r[e + 8] = hi[e]; }
  return r;
}

__device__ __forceinline__ v8f vzero8() {
  v8f z;
#pragma unroll
  for (int e = 0; e < 8; ++e) z[e] = 0.0f;
  return z;
}

#if ASYNC_LDS
typedef __attribute__((address_space(1))) i4 gi4;   // global int4
typedef __attribute__((address_space(3))) i4 li4;   // LDS int4
// async copy of 16 contiguous halfs (32B) global -> LDS, tracked by ASYNCcnt
__device__ __forceinline__ void async_cp16h(const _Float16* g, _Float16* l) {
  __builtin_amdgcn_global_load_async_to_lds_b128((gi4*)g,       (li4*)l,       0, 0);
  __builtin_amdgcn_global_load_async_to_lds_b128((gi4*)(g + 8), (li4*)(l + 8), 0, 0);
}
#endif

// ---------------------------------------------------------------------------
// Kernel 1: Q/K/V projections.  Y = X @ W^T + b  (f32 in, f16 out)
// Double-buffered LDS, register-staged global loads one k-step ahead.
// Q,K stored [head][s][64]; V stored transposed [head][d][s].
// ---------------------------------------------------------------------------
__global__ __launch_bounds__(256)
void qkv_proj_kernel(const float* __restrict__ Xq, const float* __restrict__ Xk,
                     const float* __restrict__ Xv,
                     const float* __restrict__ Wq, const float* __restrict__ Wk,
                     const float* __restrict__ Wv,
                     const float* __restrict__ bq, const float* __restrict__ bk,
                     const float* __restrict__ bv,
                     _Float16* __restrict__ qf, _Float16* __restrict__ kf,
                     _Float16* __restrict__ vT)
{
  __shared__ _Float16 As[2][128 * 40];   // [buf][row][k], stride 40 halfs
  __shared__ _Float16 Bs[2][128 * 40];

  const int z = blockIdx.z;
  const float* X    = (z == 0) ? Xq : (z == 1) ? Xk : Xv;
  const float* W    = (z == 0) ? Wq : (z == 1) ? Wk : Wv;
  const float* bias = (z == 0) ? bq : (z == 1) ? bk : bv;

  const int m0  = blockIdx.y * 128;
  const int n0  = blockIdx.x * 128;
  const int tid = threadIdx.x;
  const int lane = tid & 31;
  const int w    = tid >> 5;
  const int wr   = w >> 2;      // 0..1 : wave M offset = wr*64
  const int wc   = w & 3;       // 0..3 : wave N offset = wc*32
  const int l16  = lane & 15;
  const int lh   = lane >> 4;

  v8f acc[4][2];
#pragma unroll
  for (int mt = 0; mt < 4; ++mt)
#pragma unroll
    for (int nt = 0; nt < 2; ++nt) acc[mt][nt] = vzero8();

  const int lrow = tid >> 1;          // 0..127
  const int lcg  = (tid & 1) * 16;    // 0 or 16

  float ar[16], br[16];
  auto loadA = [&](int k0) {
    const float* s = X + (size_t)(m0 + lrow) * DM + k0 + lcg;
    if (k0 + 32 < DM) __builtin_prefetch(s + 32, 0, 1);
#pragma unroll
    for (int i = 0; i < 16; ++i) ar[i] = s[i];
  };
  auto loadB = [&](int k0) {
    const float* s = W + (size_t)(n0 + lrow) * DM + k0 + lcg;
    if (k0 + 32 < DM) __builtin_prefetch(s + 32, 0, 1);
#pragma unroll
    for (int i = 0; i < 16; ++i) br[i] = s[i];
  };
  auto stage = [&](int buf) {
    _Float16* da = &As[buf][lrow * 40 + lcg];
    _Float16* db = &Bs[buf][lrow * 40 + lcg];
#pragma unroll
    for (int i = 0; i < 16; ++i) da[i] = (_Float16)ar[i];
#pragma unroll
    for (int i = 0; i < 16; ++i) db[i] = (_Float16)br[i];
  };

  loadA(0); loadB(0);
  stage(0);

  int cur = 0;
  for (int k0 = 0; k0 < DM; k0 += 32) {
    const bool hn = (k0 + 32) < DM;
    if (hn) { loadA(k0 + 32); loadB(k0 + 32); }   // overlap with compute below
    __syncthreads();                               // publish buffer `cur`

    v16h a[4], b[2];
#pragma unroll
    for (int mt = 0; mt < 4; ++mt) {
      const _Float16* ap = &As[cur][(wr * 64 + mt * 16 + l16) * 40];
      a[mt] = cat8(*(const v8h*)(ap + 16 * lh), *(const v8h*)(ap + 16 * lh + 8));
    }
#pragma unroll
    for (int nt = 0; nt < 2; ++nt) {
      const _Float16* bp = &Bs[cur][(wc * 32 + nt * 16 + l16) * 40 + 16 * lh];
      b[nt] = cat8(*(const v8h*)(bp), *(const v8h*)(bp + 8));
    }
#pragma unroll
    for (int mt = 0; mt < 4; ++mt)
#pragma unroll
      for (int nt = 0; nt < 2; ++nt)
        acc[mt][nt] = wmma_f16(a[mt], b[nt], acc[mt][nt]);

    if (hn) stage(cur ^ 1);
    cur ^= 1;
  }

  _Float16* qkdst = (z == 0) ? qf : kf;
#pragma unroll
  for (int mt = 0; mt < 4; ++mt) {
#pragma unroll
    for (int nt = 0; nt < 2; ++nt) {
      const int n    = n0 + wc * 32 + nt * 16 + l16;
      const float bb = bias[n];
      const int head = n >> 6;
      const int d    = n & 63;
#pragma unroll
      for (int r = 0; r < 8; ++r) {
        const int srow  = m0 + wr * 64 + mt * 16 + r + 8 * lh;
        const float val = acc[mt][nt][r] + bb;
        if (z == 2) {
          vT[(size_t)head * (DKH * S_LEN) + (size_t)d * S_LEN + srow] = (_Float16)val;
        } else {
          qkdst[(size_t)head * (S_LEN * DKH) + (size_t)srow * DKH + d] = (_Float16)val;
        }
      }
    }
  }
}

// ---------------------------------------------------------------------------
// Kernel 2: sliding-window causal flash attention.
// One wave per (head, 16-row query block). 4 waves / workgroup.
// ---------------------------------------------------------------------------
__global__ __launch_bounds__(128)
void attn_kernel(const _Float16* __restrict__ qf, const _Float16* __restrict__ kf,
                 const _Float16* __restrict__ vT, _Float16* __restrict__ ctx)
{
  __shared__ _Float16 Plds[4 * 16 * 40];   // per-wave 16x32 P tile, stride 40

  const int lane = threadIdx.x & 31;
  const int wid  = threadIdx.x >> 5;
  const int l16  = lane & 15;
  const int lh   = lane >> 4;
  const int head = blockIdx.y;
  const int i0   = (blockIdx.x * 4 + wid) * 16;

  const _Float16* qh = qf + (size_t)head * S_LEN * DKH;
  const _Float16* kh = kf + (size_t)head * S_LEN * DKH;
  const _Float16* vh = vT + (size_t)head * DKH * S_LEN;
  _Float16* pw = &Plds[wid * 16 * 40];

  // Q fragments: A-matrix 16x32, two k-steps covering d=0..63
  v16h qa[2];
#pragma unroll
  for (int ks = 0; ks < 2; ++ks) {
    const _Float16* qp = qh + (size_t)(i0 + l16) * DKH + ks * 32;
    qa[ks] = cat8(*(const v8h*)(qp + 16 * lh), *(const v8h*)(qp + 16 * lh + 8));
  }

  float m[8], lsum[8];
  v8f o[4];
#pragma unroll
  for (int r = 0; r < 8; ++r) { m[r] = -1e30f; lsum[r] = 0.0f; }
#pragma unroll
  for (int nt = 0; nt < 4; ++nt) o[nt] = vzero8();

  int jstart = i0 - WINDOW;
  if (jstart < 0) jstart = 0;
  jstart &= ~31;

  for (int jb = jstart; jb <= i0 + 15; jb += 32) {
    // scores for 32 keys = two 16x16 tiles
    v8f s[2];
#pragma unroll
    for (int t = 0; t < 2; ++t) {
      v8f a = vzero8();
#pragma unroll
      for (int ks = 0; ks < 2; ++ks) {
        const _Float16* kp = kh + (size_t)(jb + t * 16 + l16) * DKH + ks * 32 + 16 * lh;
        a = wmma_f16(qa[ks], cat8(*(const v8h*)(kp), *(const v8h*)(kp + 8)), a);
      }
      s[t] = a;
    }

    // mask + scale + running row-max (row = i0 + r + 8*lh, col = jb + t*16 + l16)
    float mnew[8];
#pragma unroll
    for (int r = 0; r < 8; ++r) {
      const int irow = i0 + r + 8 * lh;
      const int j0 = jb + l16, j1 = jb + 16 + l16;
      const bool k0 = (j0 <= irow) && (irow - j0 <= WINDOW);
      const bool k1 = (j1 <= irow) && (irow - j1 <= WINDOW);
      const float v0 = k0 ? s[0][r] * 0.125f : -1e30f;
      const float v1 = k1 ? s[1][r] * 0.125f : -1e30f;
      s[0][r] = v0; s[1][r] = v1;
      float v = fmaxf(v0, v1);
      v = fmaxf(v, __shfl_xor(v, 1));
      v = fmaxf(v, __shfl_xor(v, 2));
      v = fmaxf(v, __shfl_xor(v, 4));
      v = fmaxf(v, __shfl_xor(v, 8));
      mnew[r] = fmaxf(m[r], v);
    }

    // exp, row-sum, rescale O, write P tile to LDS
#pragma unroll
    for (int r = 0; r < 8; ++r) {
      const int irow = i0 + r + 8 * lh;
      const float mn = mnew[r];
      const float cs = __expf(m[r] - mn);
      const int j0 = jb + l16, j1 = jb + 16 + l16;
      const bool k0 = (j0 <= irow) && (irow - j0 <= WINDOW);
      const bool k1 = (j1 <= irow) && (irow - j1 <= WINDOW);
      const float p0 = k0 ? __expf(s[0][r] - mn) : 0.0f;
      const float p1 = k1 ? __expf(s[1][r] - mn) : 0.0f;
      float rs = p0 + p1;
      rs += __shfl_xor(rs, 1);
      rs += __shfl_xor(rs, 2);
      rs += __shfl_xor(rs, 4);
      rs += __shfl_xor(rs, 8);
      lsum[r] = lsum[r] * cs + rs;
      m[r] = mn;
#pragma unroll
      for (int nt = 0; nt < 4; ++nt) o[nt][r] = o[nt][r] * cs;
      pw[(r + 8 * lh) * 40 + l16]      = (_Float16)p0;
      pw[(r + 8 * lh) * 40 + 16 + l16] = (_Float16)p1;
    }

    asm volatile("s_wait_dscnt 0" ::: "memory");   // wave-local LDS RAW fence

    // re-fragment P as A-matrix (16x32) from LDS
    v16h pa;
    {
      const _Float16* pp = &pw[l16 * 40];
      pa = cat8(*(const v8h*)(pp + 16 * lh), *(const v8h*)(pp + 16 * lh + 8));
    }

    // O += P @ V  (B fragment loads contiguous keys from V^T)
#pragma unroll
    for (int nt = 0; nt < 4; ++nt) {
      const _Float16* vp = vh + (size_t)(nt * 16 + l16) * S_LEN + jb + 16 * lh;
      o[nt] = wmma_f16(pa, cat8(*(const v8h*)(vp), *(const v8h*)(vp + 8)), o[nt]);
    }
  }

  // normalize and write context [s][1024] f16
#pragma unroll
  for (int r = 0; r < 8; ++r) {
    const float inv = 1.0f / lsum[r];
    const int irow  = i0 + r + 8 * lh;
#pragma unroll
    for (int nt = 0; nt < 4; ++nt) {
      ctx[(size_t)irow * DM + head * DKH + nt * 16 + l16] =
          (_Float16)(o[nt][r] * inv);
    }
  }
}

// ---------------------------------------------------------------------------
// Kernel 3: output projection.  out = ctx @ Wo^T + bo  (f16 x f16 -> f32)
// A operand staged via GLOBAL_LOAD_ASYNC_TO_LDS (ASYNCcnt) when available.
// ---------------------------------------------------------------------------
__global__ __launch_bounds__(256)
void out_proj_kernel(const _Float16* __restrict__ ctx, const float* __restrict__ Wo,
                     const float* __restrict__ bo, float* __restrict__ out)
{
  __shared__ _Float16 As[2][128 * 40];
  __shared__ _Float16 Bs[2][128 * 40];

  const int m0  = blockIdx.y * 128;
  const int n0  = blockIdx.x * 128;
  const int tid = threadIdx.x;
  const int lane = tid & 31;
  const int w    = tid >> 5;
  const int wr   = w >> 2;
  const int wc   = w & 3;
  const int l16  = lane & 15;
  const int lh   = lane >> 4;

  v8f acc[4][2];
#pragma unroll
  for (int mt = 0; mt < 4; ++mt)
#pragma unroll
    for (int nt = 0; nt < 2; ++nt) acc[mt][nt] = vzero8();

  const int lrow = tid >> 1;
  const int lcg  = (tid & 1) * 16;

  float br[16];
#if !ASYNC_LDS
  _Float16 ah[16];
#endif

  auto loadA = [&](int k0, int buf) {
    const _Float16* s = ctx + (size_t)(m0 + lrow) * DM + k0 + lcg;
#if ASYNC_LDS
    async_cp16h(s, &As[buf][lrow * 40 + lcg]);
#else
    (void)buf;
#pragma unroll
    for (int i = 0; i < 16; ++i) ah[i] = s[i];
#endif
  };
  auto loadB = [&](int k0) {
    const float* s = Wo + (size_t)(n0 + lrow) * DM + k0 + lcg;
    if (k0 + 32 < DM) __builtin_prefetch(s + 32, 0, 1);
#pragma unroll
    for (int i = 0; i < 16; ++i) br[i] = s[i];
  };
  auto stage = [&](int buf) {
#if !ASYNC_LDS
    _Float16* da = &As[buf][lrow * 40 + lcg];
#pragma unroll
    for (int i = 0; i < 16; ++i) da[i] = ah[i];
#endif
    _Float16* db = &Bs[buf][lrow * 40 + lcg];
#pragma unroll
    for (int i = 0; i < 16; ++i) db[i] = (_Float16)br[i];
  };

  loadA(0, 0); loadB(0);
  stage(0);

  int cur = 0;
  for (int k0 = 0; k0 < DM; k0 += 32) {
    const bool hn = (k0 + 32) < DM;
    if (hn) { loadA(k0 + 32, cur ^ 1); loadB(k0 + 32); }
#if ASYNC_LDS
    if (hn) {
      asm volatile("s_wait_asynccnt 0x2" ::: "memory");  // buffer `cur` landed
    } else {
      asm volatile("s_wait_asynccnt 0x0" ::: "memory");
    }
#endif
    __syncthreads();

    v16h a[4], b[2];
#pragma unroll
    for (int mt = 0; mt < 4; ++mt) {
      const _Float16* ap = &As[cur][(wr * 64 + mt * 16 + l16) * 40];
      a[mt] = cat8(*(const v8h*)(ap + 16 * lh), *(const v8h*)(ap + 16 * lh + 8));
    }
#pragma unroll
    for (int nt = 0; nt < 2; ++nt) {
      const _Float16* bp = &Bs[cur][(wc * 32 + nt * 16 + l16) * 40 + 16 * lh];
      b[nt] = cat8(*(const v8h*)(bp), *(const v8h*)(bp + 8));
    }
#pragma unroll
    for (int mt = 0; mt < 4; ++mt)
#pragma unroll
      for (int nt = 0; nt < 2; ++nt)
        acc[mt][nt] = wmma_f16(a[mt], b[nt], acc[mt][nt]);

    if (hn) stage(cur ^ 1);
    cur ^= 1;
  }

#pragma unroll
  for (int mt = 0; mt < 4; ++mt) {
#pragma unroll
    for (int nt = 0; nt < 2; ++nt) {
      const int n = n0 + wc * 32 + nt * 16 + l16;
      const float bb = bo[n];
#pragma unroll
      for (int r = 0; r < 8; ++r) {
        const int srow = m0 + wr * 64 + mt * 16 + r + 8 * lh;
        out[(size_t)srow * DM + n] = acc[mt][nt][r] + bb;
      }
    }
  }
}

// ---------------------------------------------------------------------------
extern "C" void kernel_launch(void* const* d_in, const int* in_sizes, int n_in,
                              void* d_out, int out_size, void* d_ws, size_t ws_size,
                              hipStream_t stream) {
  (void)in_sizes; (void)n_in; (void)out_size; (void)ws_size;
  const float* Xq = (const float*)d_in[0];
  const float* Xk = (const float*)d_in[1];
  const float* Xv = (const float*)d_in[2];
  const float* Wq = (const float*)d_in[3];
  const float* bq = (const float*)d_in[4];
  const float* Wk = (const float*)d_in[5];
  const float* bk = (const float*)d_in[6];
  const float* Wv = (const float*)d_in[7];
  const float* bv = (const float*)d_in[8];
  const float* Wo = (const float*)d_in[9];
  const float* bo = (const float*)d_in[10];
  float* out = (float*)d_out;

  _Float16* qf  = (_Float16*)d_ws;
  _Float16* kf  = qf + (size_t)S_LEN * DM;
  _Float16* vT  = kf + (size_t)S_LEN * DM;
  _Float16* ctx = vT + (size_t)S_LEN * DM;

  dim3 g1(DM / 128, S_LEN / 128, 3);
  qkv_proj_kernel<<<g1, 256, 0, stream>>>(Xq, Xk, Xv, Wq, Wk, Wv, bq, bk, bv,
                                          qf, kf, vT);

  dim3 g2(S_LEN / 64, NH, 1);
  attn_kernel<<<g2, 128, 0, stream>>>(qf, kf, vT, ctx);

  dim3 g3(DM / 128, S_LEN / 128, 1);
  out_proj_kernel<<<g3, 256, 0, stream>>>(ctx, Wo, bo, out);
}
